// MultiAttention_45380624449560
// MI455X (gfx1250) — compile-verified, matrix operand order checked by reading, not run on hardware
//
#include <hip/hip_runtime.h>
#include <hip/hip_bf16.h>
#include <math.h>

typedef __attribute__((ext_vector_type(16))) __bf16 v16bf;
typedef __attribute__((ext_vector_type(8)))  __bf16 v8bf;
typedef __attribute__((ext_vector_type(8)))  float  v8f;
typedef __attribute__((ext_vector_type(4)))  unsigned int u32x4;
typedef __attribute__((ext_vector_type(8)))  int i32x8;
typedef __attribute__((ext_vector_type(4)))  int i32x4;

union BF16x16 { v16bf v; v8bf h[2]; };

#define B_   2
#define T_   2048
#define DM_  1024
#define H_   16
#define HKV_ 4
#define DH_  64
#define LAT_ 64
#define DFF_ 4096

#if defined(__has_builtin)
# if __has_builtin(__builtin_amdgcn_tensor_load_to_lds)
#  define HAVE_TDM 1
# endif
#endif
#ifndef HAVE_TDM
# define HAVE_TDM 0
#endif

#if HAVE_TDM
// 2D TDM copy global(bf16) -> LDS with row padding added in-flight.
// dims/strides in elements (data_size = 2B). Per CDNA5 ISA D# spec.
__device__ __forceinline__ void tdm_load_2d(const void* gptr, unsigned lds_off,
    unsigned dim0, unsigned dim1, unsigned long long stride0,
    unsigned tile0, unsigned tile1, unsigned pad_iv, unsigned pad_amt)
{
  unsigned long long ga = (unsigned long long)gptr;
  u32x4 g0;
  g0[0] = 1u;                                        // count=1, user mode
  g0[1] = lds_off;                                   // lds_addr (bytes)
  g0[2] = (unsigned)(ga & 0xFFFFFFFFu);              // global_addr[31:0]
  g0[3] = (unsigned)((ga >> 32) & 0x1FFFFFFu)        // global_addr[56:32]
        | 0x80000000u;                               // type=2 ("image")
  i32x8 g1;
  g1[0] = (int)((1u << 16)                           // data_size = 2 bytes
              | (1u << 20)                           // pad_enable
              | (pad_iv  << 22)                      // pad interval (8<<iv B)
              | (pad_amt << 25));                    // pad amount ((amt+1) DW)
  g1[1] = (int)((dim0 & 0xFFFFu) << 16);             // tensor_dim0[15:0]
  g1[2] = (int)((dim0 >> 16) | ((dim1 & 0xFFFFu) << 16));
  g1[3] = (int)((dim1 >> 16) | (tile0 << 16));       // tile_dim0
  g1[4] = (int)(tile1 & 0xFFFFu);                    // tile_dim1 (tile_dim2=0)
  g1[5] = (int)(unsigned)(stride0 & 0xFFFFFFFFull);  // tensor_dim0_stride
  g1[6] = (int)(unsigned)((stride0 >> 32) & 0xFFFFull);
  g1[7] = 0;
  i32x4 z4 = {0, 0, 0, 0};
#if __clang_major__ >= 23
  i32x8 z8 = {0, 0, 0, 0, 0, 0, 0, 0};
  __builtin_amdgcn_tensor_load_to_lds(g0, g1, z4, z4, z8, 0);
#else
  __builtin_amdgcn_tensor_load_to_lds(g0, g1, z4, z4, 0);
#endif
}
#endif

// ---------------------------------------------------------------------------
// Tiled GEMM, all-bf16 operands: C[M,N] = act(A[M,K] @ Bt[N,K]^T).
// Double-buffered TDM pipeline: tile i+1 DMA overlaps tile i WMMA; the
// issuing waves use s_wait_tensorcnt<=1 so one copy is always in flight.
// ACT: 0=none,1=silu,2=sigmoid-gate(*mul).
// OUT: 0=f32, 1=bf16, 2=bf16 V-transposed ([b][kvh][dh][t]).
// Requires M%128==0, K%32==0. N guarded (TDM zero-fills OOB rows).
// ---------------------------------------------------------------------------
template<int ACT, int OUT>
__global__ __launch_bounds__(256)
void gemm_bf16(const __bf16* __restrict__ A, const __bf16* __restrict__ Bt,
               void* __restrict__ Cout, const float* __restrict__ bias,
               const __bf16* __restrict__ mul, int M, int N, int K)
{
  constexpr int BM = 128, BN = 128, BK = 32;
  __shared__ alignas(16) __bf16 sA [2][BM][BK + 8];   // 64B rows + 16B pad
  __shared__ alignas(16) __bf16 sBt[2][BN][BK + 8];

  const int tid  = threadIdx.x;
  const int lane = tid & 31;
  const int w    = tid >> 5;      // 8 waves
  const int wr   = w >> 2;        // 0..1 : 64 rows each
  const int wc   = w & 3;         // 0..3 : 32 cols each
  const int m0   = blockIdx.y * BM;
  const int n0   = blockIdx.x * BN;
  const int half = lane >> 4;
  const int l15  = lane & 15;
  const int kofs = half * 8;

  v8f acc[4][2] = {};
  const int niter = K / BK;

#if HAVE_TDM
  // prologue: stage tile 0 into buffer 0
  if (w == 0)
    tdm_load_2d(&A[(size_t)m0 * K], (unsigned)(size_t)&sA[0][0][0],
                32u, 128u, (unsigned long long)K, 32u, 128u, 3u, 3u);
  else if (w == 1)
    tdm_load_2d(&Bt[(size_t)n0 * K], (unsigned)(size_t)&sBt[0][0][0],
                32u, (unsigned)(N - n0), (unsigned long long)K,
                32u, 128u, 3u, 3u);
#endif

  for (int i = 0; i < niter; ++i) {
    const int k0  = i * BK;
    const int buf = i & 1;
#if HAVE_TDM
    if (i + 1 < niter) {            // issue next tile into the other buffer
      if (w == 0)
        tdm_load_2d(&A[(size_t)m0 * K + k0 + BK],
                    (unsigned)(size_t)&sA[buf ^ 1][0][0],
                    32u, 128u, (unsigned long long)K, 32u, 128u, 3u, 3u);
      else if (w == 1)
        tdm_load_2d(&Bt[(size_t)n0 * K + k0 + BK],
                    (unsigned)(size_t)&sBt[buf ^ 1][0][0],
                    32u, (unsigned)(N - n0), (unsigned long long)K,
                    32u, 128u, 3u, 3u);
    }
    if (w <= 1) {                   // current tile landed; next still in flight
      if (i + 1 < niter) __builtin_amdgcn_s_wait_tensorcnt(1);
      else               __builtin_amdgcn_s_wait_tensorcnt(0);
    }
    __syncthreads();
#else
    __syncthreads();
    #pragma unroll
    for (int it = 0; it < 2; ++it) {         // 512 16B chunks / 256 threads
      int idx = tid + it * 256;
      int r = idx >> 2, c8 = (idx & 3) << 3;
      *reinterpret_cast<v8bf*>(&sA[buf][r][c8]) =
          *reinterpret_cast<const v8bf*>(&A[(size_t)(m0 + r) * K + k0 + c8]);
    }
    #pragma unroll
    for (int it = 0; it < 2; ++it) {
      int idx = tid + it * 256;
      int n = idx >> 2, c8 = (idx & 3) << 3;
      v8bf val = {};
      if (n0 + n < N)
        val = *reinterpret_cast<const v8bf*>(&Bt[(size_t)(n0 + n) * K + k0 + c8]);
      *reinterpret_cast<v8bf*>(&sBt[buf][n][c8]) = val;
    }
    __syncthreads();
#endif

    v16bf afr[4], bfr[2];
    #pragma unroll
    for (int mt = 0; mt < 4; ++mt) {
      int rr = wr * 64 + mt * 16 + l15;
      BF16x16 u;
      u.h[0] = *reinterpret_cast<const v8bf*>(&sA[buf][rr][kofs]);
      u.h[1] = *reinterpret_cast<const v8bf*>(&sA[buf][rr][16 + kofs]);
      afr[mt] = u.v;
    }
    #pragma unroll
    for (int nt = 0; nt < 2; ++nt) {
      int cc = wc * 32 + nt * 16 + l15;
      BF16x16 u;
      u.h[0] = *reinterpret_cast<const v8bf*>(&sBt[buf][cc][kofs]);
      u.h[1] = *reinterpret_cast<const v8bf*>(&sBt[buf][cc][16 + kofs]);
      bfr[nt] = u.v;
    }
    #pragma unroll
    for (int mt = 0; mt < 4; ++mt)
      #pragma unroll
      for (int nt = 0; nt < 2; ++nt)
        acc[mt][nt] = __builtin_amdgcn_wmma_f32_16x16x32_bf16(
            false, afr[mt], false, bfr[nt], (short)0, acc[mt][nt], false, false);
#if HAVE_TDM
    __syncthreads();   // reads of buf done before TDM overwrites it next iter
#endif
  }

  #pragma unroll
  for (int mt = 0; mt < 4; ++mt) {
    #pragma unroll
    for (int nt = 0; nt < 2; ++nt) {
      #pragma unroll
      for (int r = 0; r < 8; ++r) {
        int m = m0 + wr * 64 + mt * 16 + r + half * 8;
        int n = n0 + wc * 32 + nt * 16 + l15;
        if (n < N) {
          float c = acc[mt][nt][r];
          if (ACT == 1) {
            c = c / (1.0f + __expf(-c));                       // silu
          } else if (ACT == 2) {
            float g = 1.0f / (1.0f + __expf(-(c + bias[n])));  // sigmoid gate
            c = g * (float)mul[(size_t)m * N + n];
          }
          if (OUT == 0) {
            ((float*)Cout)[(size_t)m * N + n] = c;
          } else if (OUT == 1) {
            ((__bf16*)Cout)[(size_t)m * N + n] = (__bf16)c;
          } else {  // OUT == 2 : V^T layout [b][kvh][dh][t]
            int bb = m >> 11, t = m & (T_ - 1);
            int kvh = n >> 6, dh = n & 63;
            ((__bf16*)Cout)[((size_t)(bb * HKV_ + kvh) * DH_ + dh) * T_ + t] =
                (__bf16)c;
          }
        }
      }
    }
  }
}

// ---------------------------------------------------------------------------
// f32 -> bf16 conversion; and f32 [K][N] -> bf16 [N][K] transpose-convert.
// ---------------------------------------------------------------------------
__global__ void cvt_bf16(const float* __restrict__ in, __bf16* __restrict__ out,
                         int n)
{
  int i = blockIdx.x * blockDim.x + threadIdx.x;
  if (i < n) out[i] = (__bf16)in[i];
}

__global__ void transpose_cvt(const float* __restrict__ in,
                              __bf16* __restrict__ out, int K, int N)
{
  int i = blockIdx.x * blockDim.x + threadIdx.x;
  if (i < K * N) {
    int k = i / N, n = i % N;
    out[(size_t)n * K + k] = (__bf16)in[i];
  }
}

// ---------------------------------------------------------------------------
// RoPE (in-place, bf16), NH heads, DH=64; output scale folds the softmax
// 1/sqrt(DH) into q so attention does no per-element scaling.
// ---------------------------------------------------------------------------
__global__ void rope_kernel(__bf16* __restrict__ p, int NH, float scale)
{
  int idx = blockIdx.x * blockDim.x + threadIdx.x;
  int total = B_ * T_ * NH * 32;
  if (idx >= total) return;
  int i    = idx & 31;
  int head = (idx >> 5) % NH;
  int t    = (idx / (32 * NH)) % T_;
  int b    = idx / (32 * NH * T_);
  size_t base = (((size_t)b * T_ + t) * NH + head) * DH_;
  float x1 = (float)p[base + i], x2 = (float)p[base + 32 + i];
  float inv = __powf(10000.0f, -(float)i / 32.0f);
  float ang = (float)t * inv;
  float s, c;
  __sincosf(ang, &s, &c);
  p[base + i]      = (__bf16)((x1 * c - x2 * s) * scale);
  p[base + 32 + i] = (__bf16)((x2 * c + x1 * s) * scale);
}

// ---------------------------------------------------------------------------
// Flash attention, causal, GQA. One block = (b, h, 64 q rows), 4 waves x 16
// rows. K tiles and pre-transposed V tiles arrive via double-buffered TDM.
// ---------------------------------------------------------------------------
__global__ __launch_bounds__(128)
void flash_attn(const __bf16* __restrict__ q, const __bf16* __restrict__ k,
                const __bf16* __restrict__ vT, __bf16* __restrict__ o)
{
  __shared__ alignas(16) __bf16 sK [2][64][DH_ + 8];   // [key][dh]
  __shared__ alignas(16) __bf16 sVt[2][DH_][64 + 8];   // [dh][key]
  __shared__ alignas(16) __bf16 sP [4][16][64 + 8];    // wave-private P

  const int tid  = threadIdx.x;
  const int lane = tid & 31;
  const int w    = tid >> 5;
  const int half = lane >> 4;
  const int l15  = lane & 15;
  const int kofs = half * 8;

  const int qt    = blockIdx.x;
  const int h     = blockIdx.y;
  const int b     = blockIdx.z;
  const int kvh   = h >> 2;
  const int qbase = qt * 64;

  const __bf16* kbase = &k[((size_t)b * T_ * HKV_ + kvh) * DH_];
  const __bf16* vbase = &vT[((size_t)(b * HKV_ + kvh) * DH_) * T_];

  // Q fragments (A layout), straight 16B bf16 global loads
  const int qrowA = qbase + w * 16 + l15;
  const __bf16* qp = &q[(((size_t)b * T_ + qrowA) * H_ + h) * DH_];
  v16bf aq[2];
  #pragma unroll
  for (int c = 0; c < 2; ++c) {
    BF16x16 u;
    u.h[0] = *reinterpret_cast<const v8bf*>(&qp[c * 32 + kofs]);
    u.h[1] = *reinterpret_cast<const v8bf*>(&qp[c * 32 + 16 + kofs]);
    aq[c] = u.v;
  }

  v8f   oacc[4] = {};
  float mrow[8], lrow[8];
  #pragma unroll
  for (int r = 0; r < 8; ++r) { mrow[r] = -__builtin_inff(); lrow[r] = 0.0f; }

#if HAVE_TDM
  // prologue: stage key-tile 0 into buffer 0
  if (w == 0)
    tdm_load_2d(kbase, (unsigned)(size_t)&sK[0][0][0],
                64u, 64u, (unsigned long long)(HKV_ * DH_), 64u, 64u, 4u, 3u);
  else if (w == 1)
    tdm_load_2d(vbase, (unsigned)(size_t)&sVt[0][0][0],
                64u, 64u, (unsigned long long)T_, 64u, 64u, 4u, 3u);
#endif

  for (int kt = 0; kt <= qt; ++kt) {
    const int kb  = kt * 64;
    const int buf = kt & 1;
#if HAVE_TDM
    if (kt + 1 <= qt) {             // issue next key tile into other buffer
      if (w == 0)
        tdm_load_2d(&kbase[(size_t)(kb + 64) * HKV_ * DH_],
                    (unsigned)(size_t)&sK[buf ^ 1][0][0],
                    64u, 64u, (unsigned long long)(HKV_ * DH_),
                    64u, 64u, 4u, 3u);
      else if (w == 1)
        tdm_load_2d(&vbase[kb + 64],
                    (unsigned)(size_t)&sVt[buf ^ 1][0][0],
                    64u, 64u, (unsigned long long)T_, 64u, 64u, 4u, 3u);
    }
    if (w <= 1) {
      if (kt + 1 <= qt) __builtin_amdgcn_s_wait_tensorcnt(1);
      else              __builtin_amdgcn_s_wait_tensorcnt(0);
    }
    __syncthreads();
#else
    __syncthreads();
    #pragma unroll
    for (int it = 0; it < 4; ++it) {        // 512 16B chunks / 128 threads
      int idx = tid + it * 128;
      int rr = idx >> 3, c8 = (idx & 7) << 3;
      *reinterpret_cast<v8bf*>(&sK[buf][rr][c8]) =
          *reinterpret_cast<const v8bf*>(&kbase[(size_t)(kb + rr) * HKV_ * DH_ + c8]);
      *reinterpret_cast<v8bf*>(&sVt[buf][rr][c8]) =
          *reinterpret_cast<const v8bf*>(&vbase[(size_t)rr * T_ + kb + c8]);
    }
    __syncthreads();
#endif

    // S = Q K^T
    v8f sfr[4] = {};
    #pragma unroll
    for (int f = 0; f < 4; ++f) {
      int key = f * 16 + l15;
      #pragma unroll
      for (int c = 0; c < 2; ++c) {
        BF16x16 u;
        u.h[0] = *reinterpret_cast<const v8bf*>(&sK[buf][key][c * 32 + kofs]);
        u.h[1] = *reinterpret_cast<const v8bf*>(&sK[buf][key][c * 32 + 16 + kofs]);
        sfr[f] = __builtin_amdgcn_wmma_f32_16x16x32_bf16(
            false, aq[c], false, u.v, (short)0, sfr[f], false, false);
      }
    }

    // causal mask (diagonal tile only)
    if (kt == qt) {
      #pragma unroll
      for (int f = 0; f < 4; ++f) {
        int key = kb + f * 16 + l15;
        #pragma unroll
        for (int r = 0; r < 8; ++r) {
          int row = qbase + w * 16 + r + half * 8;
          if (key > row) sfr[f][r] = -__builtin_inff();
        }
      }
    }

    // online softmax: local fragment reduce, then 4-step butterfly
    #pragma unroll
    for (int r = 0; r < 8; ++r) {
      float tmax = fmaxf(fmaxf(sfr[0][r], sfr[1][r]),
                         fmaxf(sfr[2][r], sfr[3][r]));
      tmax = fmaxf(tmax, __shfl_xor(tmax, 1, 32));
      tmax = fmaxf(tmax, __shfl_xor(tmax, 2, 32));
      tmax = fmaxf(tmax, __shfl_xor(tmax, 4, 32));
      tmax = fmaxf(tmax, __shfl_xor(tmax, 8, 32));
      float mnew  = fmaxf(mrow[r], tmax);
      float alpha = __expf(mrow[r] - mnew);
      mrow[r] = mnew;
      float ps = 0.0f;
      #pragma unroll
      for (int f = 0; f < 4; ++f) {
        float p = __expf(sfr[f][r] - mnew);
        sfr[f][r] = p;
        ps += p;
      }
      ps += __shfl_xor(ps, 1, 32);
      ps += __shfl_xor(ps, 2, 32);
      ps += __shfl_xor(ps, 4, 32);
      ps += __shfl_xor(ps, 8, 32);
      lrow[r] = lrow[r] * alpha + ps;
      #pragma unroll
      for (int f = 0; f < 4; ++f) oacc[f][r] *= alpha;
    }

    // P: C layout -> LDS -> A layout (wave-private)
    #pragma unroll
    for (int f = 0; f < 4; ++f)
      #pragma unroll
      for (int r = 0; r < 8; ++r)
        sP[w][r + half * 8][f * 16 + l15] = (__bf16)sfr[f][r];
    __builtin_amdgcn_wave_barrier();

    v16bf ap[2];
    #pragma unroll
    for (int c = 0; c < 2; ++c) {
      BF16x16 u;
      u.h[0] = *reinterpret_cast<const v8bf*>(&sP[w][l15][c * 32 + kofs]);
      u.h[1] = *reinterpret_cast<const v8bf*>(&sP[w][l15][c * 32 + 16 + kofs]);
      ap[c] = u.v;
    }
    // O += P V
    #pragma unroll
    for (int f = 0; f < 4; ++f) {
      int dh = f * 16 + l15;
      #pragma unroll
      for (int c = 0; c < 2; ++c) {
        BF16x16 u;
        u.h[0] = *reinterpret_cast<const v8bf*>(&sVt[buf][dh][c * 32 + kofs]);
        u.h[1] = *reinterpret_cast<const v8bf*>(&sVt[buf][dh][c * 32 + 16 + kofs]);
        oacc[f] = __builtin_amdgcn_wmma_f32_16x16x32_bf16(
            false, ap[c], false, u.v, (short)0, oacc[f], false, false);
      }
    }
#if HAVE_TDM
    __syncthreads();   // reads of buf done before TDM overwrites it next iter
#endif
  }

  // normalize + store bf16 (b, t, h, d)
  #pragma unroll
  for (int f = 0; f < 4; ++f) {
    #pragma unroll
    for (int r = 0; r < 8; ++r) {
      int row = qbase + w * 16 + r + half * 8;
      o[(((size_t)b * T_ + row) * H_ + h) * DH_ + f * 16 + l15] =
          (__bf16)(oacc[f][r] / lrow[r]);
    }
  }
}

// ---------------------------------------------------------------------------
extern "C" void kernel_launch(void* const* d_in, const int* in_sizes, int n_in,
                              void* d_out, int out_size, void* d_ws, size_t ws_size,
                              hipStream_t stream)
{
  const float* x    = (const float*)d_in[0];
  const float* Wq   = (const float*)d_in[1];
  const float* Wkv  = (const float*)d_in[2];
  const float* Wk   = (const float*)d_in[3];
  const float* Wv   = (const float*)d_in[4];
  const float* Wg   = (const float*)d_in[5];
  const float* bg   = (const float*)d_in[6];
  const float* Wo1  = (const float*)d_in[7];
  const float* Wo2  = (const float*)d_in[8];
  float* out = (float*)d_out;

  const int BT = B_ * T_;               // 4096
  char* ws = (char*)d_ws;
  auto alloc_bf = [&](size_t elems) {
    __bf16* p = (__bf16*)ws; ws += elems * sizeof(__bf16); return p;
  };
  __bf16* xb   = alloc_bf((size_t)BT * DM_);
  __bf16* WqT  = alloc_bf((size_t)DM_ * DM_);
  __bf16* WkvT = alloc_bf((size_t)DM_ * LAT_);
  __bf16* WkT  = alloc_bf((size_t)LAT_ * HKV_ * DH_);
  __bf16* WvT  = alloc_bf((size_t)LAT_ * HKV_ * DH_);
  __bf16* WgT  = alloc_bf((size_t)DH_ * DH_);
  __bf16* Wo1T = alloc_bf((size_t)DM_ * DFF_);
  __bf16* Wo2T = alloc_bf((size_t)DFF_ * DM_);
  __bf16* qb   = alloc_bf((size_t)BT * DM_);
  __bf16* latb = alloc_bf((size_t)BT * LAT_);
  __bf16* kbuf = alloc_bf((size_t)BT * HKV_ * DH_);
  __bf16* vTb  = alloc_bf((size_t)BT * HKV_ * DH_);
  __bf16* ob   = alloc_bf((size_t)BT * DM_);
  __bf16* gbuf = alloc_bf((size_t)BT * DM_);
  __bf16* hbuf = alloc_bf((size_t)BT * DFF_);

  auto g1 = [](int n) { return dim3((n + 255) / 256); };

  // one-time bf16 conversion / weight transposition ([K][N] -> [N][K])
  cvt_bf16<<<g1(BT * DM_), 256, 0, stream>>>(x, xb, BT * DM_);
  transpose_cvt<<<g1(DM_ * DM_), 256, 0, stream>>>(Wq,  WqT,  DM_, DM_);
  transpose_cvt<<<g1(DM_ * LAT_), 256, 0, stream>>>(Wkv, WkvT, DM_, LAT_);
  transpose_cvt<<<g1(LAT_ * HKV_ * DH_), 256, 0, stream>>>(Wk, WkT, LAT_, HKV_ * DH_);
  transpose_cvt<<<g1(LAT_ * HKV_ * DH_), 256, 0, stream>>>(Wv, WvT, LAT_, HKV_ * DH_);
  transpose_cvt<<<g1(DH_ * DH_), 256, 0, stream>>>(Wg, WgT, DH_, DH_);
  transpose_cvt<<<g1(DM_ * DFF_), 256, 0, stream>>>(Wo1, Wo1T, DM_, DFF_);
  transpose_cvt<<<g1(DFF_ * DM_), 256, 0, stream>>>(Wo2, Wo2T, DFF_, DM_);

  // q = x @ Wq
  gemm_bf16<0, 1><<<dim3(DM_ / 128, BT / 128), 256, 0, stream>>>(
      xb, WqT, qb, nullptr, nullptr, BT, DM_, DM_);
  // lat = x @ Wkv_down
  gemm_bf16<0, 1><<<dim3(1, BT / 128), 256, 0, stream>>>(
      xb, WkvT, latb, nullptr, nullptr, BT, LAT_, DM_);
  // k = lat @ Wk_up
  gemm_bf16<0, 1><<<dim3(2, BT / 128), 256, 0, stream>>>(
      latb, WkT, kbuf, nullptr, nullptr, BT, HKV_ * DH_, LAT_);
  // v = lat @ Wv_up, stored transposed [b][kvh][dh][t]
  gemm_bf16<0, 2><<<dim3(2, BT / 128), 256, 0, stream>>>(
      latb, WvT, vTb, nullptr, nullptr, BT, HKV_ * DH_, LAT_);
  // RoPE (q gets the softmax scale folded in)
  rope_kernel<<<g1(B_ * T_ * H_ * 32), 256, 0, stream>>>(qb, H_, 0.125f);
  rope_kernel<<<g1(B_ * T_ * HKV_ * 32), 256, 0, stream>>>(kbuf, HKV_, 1.0f);
  // causal GQA flash attention
  flash_attn<<<dim3(T_ / 64, H_, B_), 128, 0, stream>>>(qb, kbuf, vTb, ob);
  // gate: gbuf = sigmoid(ob @ Wgate + bgate) * ob
  gemm_bf16<2, 1><<<dim3(1, (BT * H_) / 128), 256, 0, stream>>>(
      ob, WgT, gbuf, bg, ob, BT * H_, DH_, DH_);
  // h = silu(gbuf @ Wo1)
  gemm_bf16<1, 1><<<dim3(DFF_ / 128, BT / 128), 256, 0, stream>>>(
      gbuf, Wo1T, hbuf, nullptr, nullptr, BT, DFF_, DM_);
  // out = h @ Wo2  (f32)
  gemm_bf16<0, 0><<<dim3(DM_ / 128, BT / 128), 256, 0, stream>>>(
      hbuf, Wo2T, out, nullptr, nullptr, BT, DM_, DFF_);
}
